// ResAttBlock_25331717112302
// MI455X (gfx1250) — compile-verified, matrix operand order checked by reading, not run on hardware
//
#include <hip/hip_runtime.h>
#include <hip/hip_bf16.h>
#include <math.h>

// ---------------------------------------------------------------------------
// Problem constants (B=4, V=2, L=8192, D=768, H=12, W=512)
// ---------------------------------------------------------------------------
#define L_SEQ   8192
#define D_MODEL 768
#define N_HEAD  12
#define HEAD_D  64
#define WIN     512
#define BV      8                       // B*V
#define TOK     (BV * L_SEQ)            // 65536 token rows
#define D_FF    3072

typedef __attribute__((ext_vector_type(16))) __bf16        v16bf;
typedef __attribute__((ext_vector_type(8)))  float         v8f;
typedef __attribute__((ext_vector_type(4)))  unsigned int  v4u;
typedef __attribute__((ext_vector_type(8)))  int           v8i;
typedef __attribute__((ext_vector_type(4)))  int           v4i;

static __device__ __forceinline__ __bf16 f2bf(float x) { return (__bf16)x; }

static __device__ __forceinline__ v8f vzero8() {
  v8f z = {0.f, 0.f, 0.f, 0.f, 0.f, 0.f, 0.f, 0.f};
  return z;
}

// Load a 16x32 bf16 WMMA A/B fragment from a row-major [rows x stride] tile.
// ISA layout: lanes 0-15 -> K {0..7,16..23}, lanes 16-31 -> K {8..15,24..31}.
// Both halves are two contiguous 16B chunks -> vectorizable LDS/global loads.
static __device__ __forceinline__ v16bf load_frag(const __bf16* p0, int stride) {
  const int lane = threadIdx.x & 31;
  const __bf16* p = p0 + (lane & 15) * stride + ((lane >> 4) << 3);
  v16bf f;
#pragma unroll
  for (int i = 0; i < 8; ++i)  f[i] = p[i];        // K base .. base+7
#pragma unroll
  for (int i = 8; i < 16; ++i) f[i] = p[i + 8];    // K base+16 .. base+23
  return f;
}

// ---------------------------------------------------------------------------
// DPP16 row (16-lane) reductions: xor1, xor2 via quad_perm, then half-mirror
// and mirror. Commutative combine => full row reduction, zero LDS traffic.
// ---------------------------------------------------------------------------
template <int CTRL>
static __device__ __forceinline__ float dpp_mov_f(float x) {
  return __int_as_float(__builtin_amdgcn_update_dpp(
      0, __float_as_int(x), CTRL, 0xf, 0xf, true));
}
static __device__ __forceinline__ float row16_max(float x) {
  x = fmaxf(x, dpp_mov_f<0xB1>(x));    // quad_perm [1,0,3,2]  (xor 1)
  x = fmaxf(x, dpp_mov_f<0x4E>(x));    // quad_perm [2,3,0,1]  (xor 2)
  x = fmaxf(x, dpp_mov_f<0x141>(x));   // row_half_mirror
  x = fmaxf(x, dpp_mov_f<0x140>(x));   // row_mirror
  return x;
}
static __device__ __forceinline__ float row16_sum(float x) {
  x += dpp_mov_f<0xB1>(x);
  x += dpp_mov_f<0x4E>(x);
  x += dpp_mov_f<0x141>(x);
  x += dpp_mov_f<0x140>(x);
  return x;
}

// ---------------------------------------------------------------------------
// Tensor Data Mover: 2D tile (bf16) global -> LDS with per-row LDS padding.
// D# per CDNA5 ISA ch.8: group0 = {count, lds_addr, global_addr, type=2},
// group1 = {data_size=2B, pad cfg, dims, strides}. Issued by one wave.
// This toolchain exposes the 6-arg builtin:
//   (uint32x4 g0, int32x8 g1, int32x4 g2, int32x4 g3, int32x8 g4, i32 cpol)
// ---------------------------------------------------------------------------
#if defined(__has_builtin)
#if __has_builtin(__builtin_amdgcn_tensor_load_to_lds) && \
    __has_builtin(__builtin_amdgcn_s_wait_tensorcnt)
#define USE_TDM 1
#endif
#endif
#ifndef USE_TDM
#define USE_TDM 0
#endif

#if USE_TDM
static __device__ __forceinline__ void tdm_load_2d_bf16(
    unsigned lds_off, const __bf16* gptr,
    unsigned tile_d0 /*elems per row*/, unsigned tile_d1 /*rows*/,
    unsigned long long row_stride /*elems*/,
    unsigned pad_interval_code, unsigned pad_amount_code) {
  const unsigned long long ga = (unsigned long long)(size_t)gptr;
  // --- group 0 (128b): count=1 | lds_addr | global_addr[56:0] | type=2 ---
  v4u g0 = {1u, lds_off, (unsigned)(ga & 0xFFFFFFFFu),
            (unsigned)((ga >> 32) & 0x01FFFFFFu) | 0x80000000u};
  // --- group 1 (256b) ---
  const unsigned td0 = 1u << 30;                      // tensor_dim0 (no OOB)
  const unsigned td1 = 1u << 30;                      // tensor_dim1 (no OOB)
  const unsigned w0 = (1u << 16)                      // data_size = 2 bytes
                    | (1u << 20)                      // pad_enable
                    | (pad_interval_code << 22)
                    | (pad_amount_code << 25);
  const unsigned w1 = (td0 & 0xFFFFu) << 16;          // tensor_dim0[15:0]
  const unsigned w2 = (td0 >> 16) | ((td1 & 0xFFFFu) << 16);
  const unsigned w3 = (td1 >> 16) | (tile_d0 << 16);  // tile_dim0
  const unsigned w4 = tile_d1;                        // tile_dim1, tile_dim2=0
  const unsigned w5 = (unsigned)(row_stride & 0xFFFFFFFFull);
  const unsigned w6 = (unsigned)((row_stride >> 32) & 0xFFFFull);
  v8i g1 = {(int)w0, (int)w1, (int)w2, (int)w3,
            (int)w4, (int)w5, (int)w6, 0};
  v4i z4 = {0, 0, 0, 0};
  v8i z8 = {0, 0, 0, 0, 0, 0, 0, 0};
  __builtin_amdgcn_tensor_load_to_lds(g0, g1, z4, z4, z8, 0);
}
#endif

// ---------------------------------------------------------------------------
// LayerNorm (fp32 math) + bf16 cast. One 768-wide row per 256-thread block.
// Optionally also emits a raw bf16 copy of the input (for the V matmul).
// ---------------------------------------------------------------------------
__global__ __launch_bounds__(256) void ln_cast_kernel(
    const float* __restrict__ x, const float* __restrict__ g,
    const float* __restrict__ b, __bf16* __restrict__ y,
    __bf16* __restrict__ xc) {
  __shared__ float red[18];
  const int tid = threadIdx.x;
  const size_t row = blockIdx.x;
  const float* xr = x + row * D_MODEL;

  float v0 = xr[tid], v1 = xr[tid + 256], v2 = xr[tid + 512];
  float s = v0 + v1 + v2;
  float q = v0 * v0 + v1 * v1 + v2 * v2;
#pragma unroll
  for (int m = 16; m >= 1; m >>= 1) {
    s += __shfl_xor(s, m, 32);
    q += __shfl_xor(q, m, 32);
  }
  if ((tid & 31) == 0) { red[tid >> 5] = s; red[8 + (tid >> 5)] = q; }
  __syncthreads();
  if (tid < 32) {
    float ts = (tid < 8) ? red[tid] : 0.f;
    float tq = (tid < 8) ? red[8 + tid] : 0.f;
#pragma unroll
    for (int m = 4; m >= 1; m >>= 1) {
      ts += __shfl_xor(ts, m, 32);
      tq += __shfl_xor(tq, m, 32);
    }
    if (tid == 0) { red[16] = ts; red[17] = tq; }
  }
  __syncthreads();
  const float mean = red[16] * (1.f / D_MODEL);
  const float var  = red[17] * (1.f / D_MODEL) - mean * mean;
  const float inv  = rsqrtf(var + 1e-5f);

  __bf16* yr = y + row * D_MODEL;
  yr[tid]       = f2bf((v0 - mean) * inv * g[tid] + b[tid]);
  yr[tid + 256] = f2bf((v1 - mean) * inv * g[tid + 256] + b[tid + 256]);
  yr[tid + 512] = f2bf((v2 - mean) * inv * g[tid + 512] + b[tid + 512]);
  if (xc) {
    __bf16* xcr = xc + row * D_MODEL;
    xcr[tid] = f2bf(v0);
    xcr[tid + 256] = f2bf(v1);
    xcr[tid + 512] = f2bf(v2);
  }
}

// ---------------------------------------------------------------------------
// fp32 -> bf16 weight cast (grid-stride).
// ---------------------------------------------------------------------------
__global__ void cast_w_kernel(const float* __restrict__ in,
                              __bf16* __restrict__ out, size_t n) {
  size_t i = (size_t)blockIdx.x * blockDim.x + threadIdx.x;
  const size_t stride = (size_t)gridDim.x * blockDim.x;
  for (; i < n; i += stride) out[i] = f2bf(in[i]);
}

// ---------------------------------------------------------------------------
// bf16 WMMA GEMM:  C[m,n] = sum_k A[m,k] * Bt[n,k]   (Bt is N x K row-major,
// which matches torch-style weight storage W[out,in] for x @ W^T).
// Block tile 128x128, K-step 32. 8 waves: 2(M) x 4(N), each wave 64x32
// -> 8 v_wmma_f32_16x16x32_bf16 per K-step.
// Epilogue MODE: 1 = bf16, head-major [h][t][d] (Q/K),  2 = bf16 transposed
// head-major [h*64+d][t] (V),  3 = f32 resid + val,  5 = bf16 fast-GELU.
// apply_roll shifts A row reads by -W/2 mod L when index is odd (V path).
// ---------------------------------------------------------------------------
#define BM 128
#define BN 128
#define BK 32
#define LDT 40                       // padded LDS row stride (bf16 elems)

template <int MODE>
__global__ __launch_bounds__(256) void gemm_bf16_kernel(
    const __bf16* __restrict__ A, const __bf16* __restrict__ Bt,
    const float* __restrict__ bias, void* outp, const float* resid,
    const int* __restrict__ idxp, int M, int N, int K, int Lattn,
    int apply_roll, float scale) {
  __shared__ __align__(16) __bf16 As[BM * LDT];
  __shared__ __align__(16) __bf16 Bs[BN * LDT];

  const int tid  = threadIdx.x;
  const int lane = tid & 31;
  const int wave = tid >> 5;
  const int m0 = blockIdx.x * BM;
  const int n0 = blockIdx.y * BN;

  int shift = 0;
  if (apply_roll && idxp) shift = (idxp[0] & 1) ? (WIN / 2) : 0;

  const int wm = (wave >> 2) * 64;   // wave M offset within block tile
  const int wn = (wave & 3) * 32;    // wave N offset within block tile

  v8f acc[4][2];
#pragma unroll
  for (int i = 0; i < 4; ++i)
#pragma unroll
    for (int j = 0; j < 2; ++j) acc[i][j] = vzero8();

  for (int k0 = 0; k0 < K; k0 += BK) {
    // Cooperative tile loads: 128 rows x 32 bf16, 16B vectors.
#pragma unroll
    for (int p = 0; p < 2; ++p) {
      const int c4 = tid + p * 256;          // 512 chunks of 8 bf16
      const int r  = c4 >> 2;
      const int c  = (c4 & 3) * 8;
      const int gr = m0 + r;
      const int sr = apply_roll ? ((gr - shift) & (Lattn - 1)) : gr;
      *(uint4*)(&As[r * LDT + c]) =
          *(const uint4*)(A + (size_t)sr * K + k0 + c);
    }
#pragma unroll
    for (int p = 0; p < 2; ++p) {
      const int c4 = tid + p * 256;
      const int r  = c4 >> 2;
      const int c  = (c4 & 3) * 8;
      *(uint4*)(&Bs[r * LDT + c]) =
          *(const uint4*)(Bt + (size_t)(n0 + r) * K + k0 + c);
    }
    __syncthreads();

    const v16bf bf0 = load_frag(&Bs[(wn + 0)  * LDT], LDT);
    const v16bf bf1 = load_frag(&Bs[(wn + 16) * LDT], LDT);
#pragma unroll
    for (int i = 0; i < 4; ++i) {
      const v16bf af = load_frag(&As[(wm + i * 16) * LDT], LDT);
      acc[i][0] = __builtin_amdgcn_wmma_f32_16x16x32_bf16(
          false, af, false, bf0, (short)0, acc[i][0], false, false);
      acc[i][1] = __builtin_amdgcn_wmma_f32_16x16x32_bf16(
          false, af, false, bf1, (short)0, acc[i][1], false, false);
    }
    __syncthreads();
  }

  // Epilogue. C/D layout: lane l, reg r -> m = r + 8*(l>=16), n = l&15.
  const int rofs = (lane >> 4) * 8;
  const int ncol = lane & 15;
#pragma unroll
  for (int i = 0; i < 4; ++i)
#pragma unroll
    for (int j = 0; j < 2; ++j)
#pragma unroll
      for (int r = 0; r < 8; ++r) {
        const int m = m0 + wm + i * 16 + r + rofs;
        const int n = n0 + wn + j * 16 + ncol;
        const float v = (acc[i][j][r] + bias[n]) * scale;
        if (MODE == 1) {                  // head-major Q/K: [h][t][64]
          ((__bf16*)outp)[(size_t)(n >> 6) * ((size_t)Lattn * HEAD_D) +
                          (size_t)m * HEAD_D + (n & 63)] = f2bf(v);
        } else if (MODE == 2) {           // transposed V: [h*64+d][t]
          ((__bf16*)outp)[(size_t)n * Lattn + m] = f2bf(v);
        } else if (MODE == 3) {           // f32 residual add
          ((float*)outp)[(size_t)m * N + n] = resid[(size_t)m * N + n] + v;
        } else if (MODE == 5) {           // fast GELU -> bf16
          const float gge = v / (1.f + __expf(-1.702f * v));
          ((__bf16*)outp)[(size_t)m * N + n] = f2bf(gge);
        }
      }
}

// ---------------------------------------------------------------------------
// Windowed cross-view attention, flash-style online softmax.
// Grid: x = win*4 + qtile(128 rows), y = head, z = bv. 8 waves x 16 q-rows.
// K/Vt chunks staged in LDS by the Tensor Data Mover (wave 0 issues, waits
// on TENSORcnt, then the workgroup barrier publishes the tile).
// Q is pre-scaled by 1/sqrt(64) in its projection; V rows were read rolled;
// the output row is written at the un-rolled position (t - shift) mod L.
// ---------------------------------------------------------------------------
#define KC  128                          // keys per LDS chunk
#define KLD 72                           // padded K-tile stride (64 + 8)
#define VLD 136                          // padded Vt-tile stride (128 + 8)
#define PLD 40                           // padded P-tile stride

__global__ __launch_bounds__(256) void attn_win_kernel(
    const __bf16* __restrict__ q_ws, const __bf16* __restrict__ k_ws,
    const __bf16* __restrict__ v_ws, __bf16* __restrict__ att,
    const int* __restrict__ idxp) {
  __shared__ __align__(16) __bf16 ksh[KC * KLD];        // [key][d]
  __shared__ __align__(16) __bf16 vsh[HEAD_D * VLD];    // [d][key]
  __shared__ __align__(16) __bf16 psh[8 * 16 * PLD];    // per-wave P 16x32

  const int tid  = threadIdx.x;
  const int lane = tid & 31;
  const int wave = tid >> 5;

  const int qt  = blockIdx.x & 3;
  const int win = blockIdx.x >> 2;
  const int h   = blockIdx.y;
  const int bv  = blockIdx.z;

  const int shift = (idxp[0] & 1) ? (WIN / 2) : 0;

  const size_t headbase = ((size_t)bv * N_HEAD + h) * L_SEQ;
  const __bf16* qbase = q_ws + (headbase + (size_t)win * WIN + qt * 128) * HEAD_D;
  const __bf16* kbase = k_ws + (headbase + (size_t)win * WIN) * HEAD_D;
  const __bf16* vbase = v_ws + ((size_t)bv * D_MODEL + h * HEAD_D) * L_SEQ +
                        (size_t)win * WIN;

  const int q0 = wave * 16;
  const v16bf qf0 = load_frag(qbase + (size_t)q0 * HEAD_D + 0,  HEAD_D);
  const v16bf qf1 = load_frag(qbase + (size_t)q0 * HEAD_D + 32, HEAD_D);

  float rmax[8], rsum[8];
  v8f o[4];
#pragma unroll
  for (int i = 0; i < 4; ++i) o[i] = vzero8();
#pragma unroll
  for (int r = 0; r < 8; ++r) { rmax[r] = -1e30f; rsum[r] = 0.f; }

  for (int kc = 0; kc < WIN; kc += KC) {
#if USE_TDM
    if (wave == 0) {
      // K chunk: 128 rows x 64 bf16, row = 32 DWORDs -> pad interval code 4,
      // pad 4 DWORDs (8 bf16) -> code 3 => LDS stride 72.
      tdm_load_2d_bf16((unsigned)(size_t)&ksh[0],
                       kbase + (size_t)kc * HEAD_D, HEAD_D, KC,
                       (unsigned long long)HEAD_D, 4u, 3u);
      // Vt chunk: 64 rows x 128 bf16, row = 64 DWORDs -> interval code 5,
      // pad 4 DWORDs -> code 3 => LDS stride 136.
      tdm_load_2d_bf16((unsigned)(size_t)&vsh[0],
                       vbase + kc, KC, HEAD_D,
                       (unsigned long long)L_SEQ, 5u, 3u);
      __builtin_amdgcn_s_wait_tensorcnt(0);
    }
#else
#pragma unroll
    for (int p = 0; p < 4; ++p) {
      const int c8 = tid + p * 256;
      const int r  = c8 >> 3;
      const int c  = (c8 & 7) * 8;
      *(uint4*)(&ksh[r * KLD + c]) =
          *(const uint4*)(kbase + (size_t)(kc + r) * HEAD_D + c);
    }
#pragma unroll
    for (int p = 0; p < 4; ++p) {
      const int c8 = tid + p * 256;
      const int r  = c8 >> 4;
      const int c  = (c8 & 15) * 8;
      *(uint4*)(&vsh[r * VLD + c]) =
          *(const uint4*)(vbase + (size_t)r * L_SEQ + kc + c);
    }
#endif
    __syncthreads();

    for (int sc = 0; sc < KC; sc += 32) {
      // S = Q (16x64) . K^T (64x32): key rows are exactly the Bt layout.
      const v16bf kf00 = load_frag(&ksh[(sc + 0)  * KLD + 0],  KLD);
      const v16bf kf01 = load_frag(&ksh[(sc + 0)  * KLD + 32], KLD);
      const v16bf kf10 = load_frag(&ksh[(sc + 16) * KLD + 0],  KLD);
      const v16bf kf11 = load_frag(&ksh[(sc + 16) * KLD + 32], KLD);
      v8f s0 = vzero8(), s1 = vzero8();
      s0 = __builtin_amdgcn_wmma_f32_16x16x32_bf16(false, qf0, false, kf00,
                                                   (short)0, s0, false, false);
      s0 = __builtin_amdgcn_wmma_f32_16x16x32_bf16(false, qf1, false, kf01,
                                                   (short)0, s0, false, false);
      s1 = __builtin_amdgcn_wmma_f32_16x16x32_bf16(false, qf0, false, kf10,
                                                   (short)0, s1, false, false);
      s1 = __builtin_amdgcn_wmma_f32_16x16x32_bf16(false, qf1, false, kf11,
                                                   (short)0, s1, false, false);

      // Online softmax across these 32 keys. Each query row lives in one
      // 16-lane half -> DPP16 row reductions (no LDS traffic).
      float p0[8], p1[8];
#pragma unroll
      for (int r = 0; r < 8; ++r) {
        const float cm   = row16_max(fmaxf(s0[r], s1[r]));
        const float nm   = fmaxf(rmax[r], cm);
        const float corr = __expf(rmax[r] - nm);
        p0[r] = __expf(s0[r] - nm);
        p1[r] = __expf(s1[r] - nm);
        const float ps = row16_sum(p0[r] + p1[r]);
        rsum[r] = rsum[r] * corr + ps;
        rmax[r] = nm;
#pragma unroll
        for (int i = 0; i < 4; ++i) o[i][r] *= corr;
      }

      // C-layout -> A-layout through per-wave LDS, then P (16x32) . V (32x64).
      __bf16* pw = &psh[wave * 16 * PLD];
#pragma unroll
      for (int r = 0; r < 8; ++r) {
        const int row = r + ((lane >> 4) << 3);
        pw[row * PLD + (lane & 15)]      = f2bf(p0[r]);
        pw[row * PLD + 16 + (lane & 15)] = f2bf(p1[r]);
      }
      asm volatile("s_wait_dscnt 0" ::: "memory");   // wave-local LDS RAW
      const v16bf pf = load_frag(pw, PLD);
#pragma unroll
      for (int i = 0; i < 4; ++i) {
        const v16bf vf = load_frag(&vsh[(i * 16) * VLD + sc], VLD);
        o[i] = __builtin_amdgcn_wmma_f32_16x16x32_bf16(
            false, pf, false, vf, (short)0, o[i], false, false);
      }
    }
    __syncthreads();
  }

  // Normalize and scatter to the shifted-back position, row-major [t][768].
#pragma unroll
  for (int r = 0; r < 8; ++r) {
    const int row  = q0 + r + ((lane >> 4) << 3);
    const int t    = win * WIN + qt * 128 + row;
    const int tout = (t - shift) & (L_SEQ - 1);
    const float inv = 1.f / rsum[r];
#pragma unroll
    for (int i = 0; i < 4; ++i) {
      const int d = i * 16 + (lane & 15);
      att[((size_t)bv * L_SEQ + tout) * D_MODEL + h * HEAD_D + d] =
          f2bf(o[i][r] * inv);
    }
  }
}

// ---------------------------------------------------------------------------
// Host orchestration.
// ---------------------------------------------------------------------------
extern "C" void kernel_launch(void* const* d_in, const int* in_sizes, int n_in,
                              void* d_out, int out_size, void* d_ws,
                              size_t ws_size, hipStream_t stream) {
  (void)in_sizes; (void)n_in; (void)out_size; (void)ws_size;

  const float* x      = (const float*)d_in[0];
  const float* ln1_g  = (const float*)d_in[1];
  const float* ln1_b  = (const float*)d_in[2];
  const float* ln2_g  = (const float*)d_in[3];
  const float* ln2_b  = (const float*)d_in[4];
  const float* a_wqkv[2] = {(const float*)d_in[5],  (const float*)d_in[9]};
  const float* a_bqkv[2] = {(const float*)d_in[6],  (const float*)d_in[10]};
  const float* a_wo[2]   = {(const float*)d_in[7],  (const float*)d_in[11]};
  const float* a_bo[2]   = {(const float*)d_in[8],  (const float*)d_in[12]};
  const float* fc_w   = (const float*)d_in[13];
  const float* fc_b   = (const float*)d_in[14];
  const float* proj_w = (const float*)d_in[15];
  const float* proj_b = (const float*)d_in[16];
  const int*   idx    = (const int*)d_in[17];
  float* out = (float*)d_out;

  // Workspace layout (bf16 elements).
  __bf16* wsb = (__bf16*)d_ws;
  const size_t TOKD = (size_t)TOK * D_MODEL;     // 50,331,648 elems per tensor
  __bf16* y_bf   = wsb;                          // LN1(x) / later LN2(x2)
  __bf16* x_bf   = wsb + TOKD;                   // bf16(x)
  __bf16* q_ws   = wsb + 2 * TOKD;               // [bv][h][t][64], *0.125
  __bf16* k_ws   = wsb + 3 * TOKD;               // [bv][h][t][64]
  __bf16* v_ws   = wsb + 4 * TOKD;               // [bv][h*64+d][t]
  __bf16* att_bf = wsb + 5 * TOKD;               // [bv][t][768]
  __bf16* h_bf   = q_ws;                         // MLP hidden overlays q/k/v/att
  __bf16* wreg   = wsb + 6 * TOKD;
  size_t o = 0;
  __bf16* wqkv_bf[2]; __bf16* wo_bf[2];
  wqkv_bf[0] = wreg + o; o += (size_t)3 * D_MODEL * D_MODEL;
  wqkv_bf[1] = wreg + o; o += (size_t)3 * D_MODEL * D_MODEL;
  wo_bf[0]   = wreg + o; o += (size_t)D_MODEL * D_MODEL;
  wo_bf[1]   = wreg + o; o += (size_t)D_MODEL * D_MODEL;
  __bf16* fc_bf   = wreg + o; o += (size_t)D_FF * D_MODEL;
  __bf16* proj_bf = wreg + o; o += (size_t)D_MODEL * D_FF;

  // 1) Weights -> bf16.
  cast_w_kernel<<<1024, 256, 0, stream>>>(a_wqkv[0], wqkv_bf[0], (size_t)3 * D_MODEL * D_MODEL);
  cast_w_kernel<<<1024, 256, 0, stream>>>(a_wqkv[1], wqkv_bf[1], (size_t)3 * D_MODEL * D_MODEL);
  cast_w_kernel<<<1024, 256, 0, stream>>>(a_wo[0],   wo_bf[0],   (size_t)D_MODEL * D_MODEL);
  cast_w_kernel<<<1024, 256, 0, stream>>>(a_wo[1],   wo_bf[1],   (size_t)D_MODEL * D_MODEL);
  cast_w_kernel<<<1024, 256, 0, stream>>>(fc_w,   fc_bf,   (size_t)D_FF * D_MODEL);
  cast_w_kernel<<<1024, 256, 0, stream>>>(proj_w, proj_bf, (size_t)D_MODEL * D_FF);

  // 2) LN1 (+ raw bf16 copy of x).
  ln_cast_kernel<<<TOK, 256, 0, stream>>>(x, ln1_g, ln1_b, y_bf, x_bf);

  // 3) QKV projections per (b, view). Q/K from (un-rolled) y, V from rolled x.
  const dim3 gq(L_SEQ / BM, D_MODEL / BN);
  for (int bv = 0; bv < BV; ++bv) {
    const int v = bv & 1;
    const __bf16* ya = y_bf + (size_t)bv * L_SEQ * D_MODEL;        // own view
    const __bf16* yo = y_bf + (size_t)(bv ^ 1) * L_SEQ * D_MODEL;  // other view
    const __bf16* xa = x_bf + (size_t)bv * L_SEQ * D_MODEL;
    gemm_bf16_kernel<1><<<gq, 256, 0, stream>>>(
        ya, wqkv_bf[v], a_bqkv[v],
        q_ws + (size_t)bv * N_HEAD * L_SEQ * HEAD_D, nullptr, idx,
        L_SEQ, D_MODEL, D_MODEL, L_SEQ, 0, 0.125f);                // 1/sqrt(64)
    gemm_bf16_kernel<1><<<gq, 256, 0, stream>>>(
        yo, wqkv_bf[v] + (size_t)D_MODEL * D_MODEL, a_bqkv[v] + D_MODEL,
        k_ws + (size_t)bv * N_HEAD * L_SEQ * HEAD_D, nullptr, idx,
        L_SEQ, D_MODEL, D_MODEL, L_SEQ, 0, 1.0f);
    gemm_bf16_kernel<2><<<gq, 256, 0, stream>>>(
        xa, wqkv_bf[v] + (size_t)2 * D_MODEL * D_MODEL, a_bqkv[v] + 2 * D_MODEL,
        v_ws + (size_t)bv * D_MODEL * L_SEQ, nullptr, idx,
        L_SEQ, D_MODEL, D_MODEL, L_SEQ, /*roll=*/1, 1.0f);
  }

  // 4) Windowed cross-view attention.
  attn_win_kernel<<<dim3(4 * (L_SEQ / WIN), N_HEAD, BV), 256, 0, stream>>>(
      q_ws, k_ws, v_ws, att_bf, idx);

  // 5) Attention output projection + residual -> d_out (fp32).
  for (int bv = 0; bv < BV; ++bv) {
    const int v = bv & 1;
    gemm_bf16_kernel<3><<<gq, 256, 0, stream>>>(
        att_bf + (size_t)bv * L_SEQ * D_MODEL, wo_bf[v], a_bo[v],
        out + (size_t)bv * L_SEQ * D_MODEL, x + (size_t)bv * L_SEQ * D_MODEL,
        idx, L_SEQ, D_MODEL, D_MODEL, L_SEQ, 0, 1.0f);
  }

  // 6) LN2 on x2 = d_out.
  ln_cast_kernel<<<TOK, 256, 0, stream>>>(out, ln2_g, ln2_b, y_bf, nullptr);

  // 7) MLP fc + fast GELU -> bf16 hidden.
  gemm_bf16_kernel<5><<<dim3(TOK / BM, D_FF / BN), 256, 0, stream>>>(
      y_bf, fc_bf, fc_b, h_bf, nullptr, idx,
      TOK, D_FF, D_MODEL, L_SEQ, 0, 1.0f);

  // 8) MLP proj + residual, in place on d_out.
  gemm_bf16_kernel<3><<<dim3(TOK / BM, D_MODEL / BN), 256, 0, stream>>>(
      h_bf, proj_bf, proj_b, out, out, idx,
      TOK, D_MODEL, D_FF, L_SEQ, 0, 1.0f);
}
